// denoiseNet_adaptive_thre6_43276090474533
// MI455X (gfx1250) — matrix operand order repaired and checked by reading, not compiled, not run on hardware
//
#include <hip/hip_runtime.h>
#include <hip/hip_bf16.h>

typedef __attribute__((ext_vector_type(16))) __bf16 v16bf;
typedef __attribute__((ext_vector_type(8)))  __bf16 v8bf;
typedef __attribute__((ext_vector_type(8)))  float  v8f;
typedef __attribute__((ext_vector_type(4)))  float  v4f;
typedef __attribute__((ext_vector_type(2)))  float  v2f;
typedef __attribute__((ext_vector_type(4)))  unsigned int v4u;
typedef __attribute__((ext_vector_type(8)))  int    v8i;
typedef __attribute__((ext_vector_type(4)))  int    v4i;

// ---------------- geometry / sizes ----------------
constexpr int TSZ = 16, STR = 4, IMG = 224, G = 53;
constexpr int LL  = G * G;                       // 2809
constexpr int C0 = 3 * LL;                       // 8427
constexpr int C1 = 3 * LL / 4;                   // 2106
constexpr int C2 = 3 * LL / 16;                  // 526
constexpr int C3 = C1, C4 = C0;
// padded dims: K dims padded to x128 (TDM K-step), M dims to x16
constexpr int C0P = 8448, C1P = 2176, C2P = 640, C3P = 2176, C4P = 8432;
constexpr int BATCH = 8;
constexpr int NCOL  = BATCH * 256;               // 2048 GEMM columns
constexpr int DSTRIDE = 8448;                    // row stride of dct/clean fp32 buffers
constexpr float PI_F = 3.14159265358979323846f;

// ---------------- WMMA helpers ----------------
__device__ inline v8f wmma_bf16x32(v16bf a, v16bf b, v8f c) {
  return __builtin_amdgcn_wmma_f32_16x16x32_bf16(false, a, false, b, (short)0, c,
                                                 false, false);
}
__device__ inline v8f wmma_f32k4(v2f a, v2f b, v8f c) {
  return __builtin_amdgcn_wmma_f32_16x16x4_f32(false, a, false, b, (short)0, c,
                                               false, false);
}

// ---------------- TDM: stage a 128-row x 256-byte 2D tile into LDS ----------
// D# per CDNA5 ISA: group0 = {count, lds_addr, gaddr57, type=2},
// group1 = {data_size=8B, tensor_dim0=32u, tensor_dim1=128, tile_dim0=32u,
//           tile_dim1=128, dim0_stride = row stride in 8B units}.
__device__ inline void tdm_stage(const __bf16* lds_dst, const __bf16* gsrc,
                                 unsigned stride8) {
  unsigned long long ga = (unsigned long long)(uintptr_t)gsrc;
  v4u g0;
  g0[0] = 1u;                                        // count = 1 valid descriptor
  g0[1] = (unsigned)(uintptr_t)(const void*)lds_dst; // LDS byte address (low 32b)
  g0[2] = (unsigned)ga;                              // global_addr[31:0]
  g0[3] = (unsigned)((ga >> 32) & 0x01FFFFFFu) | (2u << 30);  // addr[56:32]|type=2
  v8i g1;
  g1[0] = 0x00030000;          // workgroup_mask=0, data_size=3 (8 bytes)
  g1[1] = (int)(32u << 16);    // tensor_dim0 = 32 (x8B = 256B row window)
  g1[2] = (int)(128u << 16);   // tensor_dim1 = 128 rows
  g1[3] = (int)(32u << 16);    // tile_dim0 = 32 (x8B)
  g1[4] = 128;                 // tile_dim1 = 128, tile_dim2 = 0 (2D)
  g1[5] = (int)stride8;        // tensor_dim0_stride (8B units)
  g1[6] = 0; g1[7] = 0;
  v4i gz4 = {0, 0, 0, 0};
  v8i gz8 = {0, 0, 0, 0, 0, 0, 0, 0};
  // 6-arg form (clang-23 / therock-10.0 toolchain)
  __builtin_amdgcn_tensor_load_to_lds(g0, g1, gz4, gz4, gz8, 0);
}

// ---------------- tiny utility kernels ----------------
__global__ void k_init_dmat(float* d) {
  int i = threadIdx.x;               // 256 threads
  int r = i >> 4, c = i & 15;
  d[i] = 2.f * __cosf(PI_F * (float)((2 * c + 1) * r) / 32.f);
  d[256 + i] = __cosf(PI_F * (float)((2 * r + 1) * c) / 32.f) *
               ((c == 0) ? 0.5f : 1.f) * (1.f / 16.f);
}

__global__ void k_zero_u32(uint32_t* p, size_t n) {
  for (size_t i = blockIdx.x * (size_t)blockDim.x + threadIdx.x; i < n;
       i += (size_t)gridDim.x * blockDim.x)
    p[i] = 0u;
}

__global__ void k_wcast(const float* __restrict__ w, __bf16* __restrict__ dst,
                        int O, int C, int Cp, size_t total) {
  for (size_t i = blockIdx.x * (size_t)blockDim.x + threadIdx.x; i < total;
       i += (size_t)gridDim.x * blockDim.x) {
    int o = (int)(i / Cp), c = (int)(i % Cp);
    float v = (o < O && c < C) ? w[(size_t)o * C + c] : 0.f;
    dst[i] = (__bf16)v;
  }
}

// ---------------- forward DCT: tiles -> dct(fp32) + act0(bf16, (n,k) layout) --
__global__ void k_dct(const float* __restrict__ x, const float* __restrict__ dmat,
                      float* __restrict__ dct, __bf16* __restrict__ act0) {
  __shared__ float tile[8][256];
  __shared__ float ub[8][256];
  int wid = threadIdx.x >> 5, lane = threadIdx.x & 31;
  int t = blockIdx.x * 8 + wid;                  // grid = 8427 blocks, exact
  int b = t / C0, c = t % C0;
  int ch = c % 3, gi = c / 3, gx = gi % G, gy = gi / G;
  const float* src =
      x + (((size_t)(b * 3 + ch) * IMG + gy * STR) * IMG + gx * STR);
  int m = lane >> 1, j0 = (lane & 1) * 8;
  v4f r0 = *(const v4f*)(src + (size_t)m * IMG + j0);
  v4f r1 = *(const v4f*)(src + (size_t)m * IMG + j0 + 4);
  *(v4f*)&tile[wid][m * 16 + j0]     = r0;
  *(v4f*)&tile[wid][m * 16 + j0 + 4] = r1;
  __syncthreads();
  int half = lane >> 4, q = lane & 15;
  // stage A: U = T * D^T
  v8f acc = {};
#pragma unroll
  for (int kk = 0; kk < 4; ++kk) {
    int k0 = kk * 4 + half * 2;
    v2f a, bf;
    a[0]  = tile[wid][q * 16 + k0];
    a[1]  = tile[wid][q * 16 + k0 + 1];
    bf[0] = dmat[q * 16 + k0];
    bf[1] = dmat[q * 16 + k0 + 1];
    acc = wmma_f32k4(a, bf, acc);
  }
#pragma unroll
  for (int v = 0; v < 8; ++v) ub[wid][(v + 8 * half) * 16 + q] = acc[v];
  __syncthreads();
  // stage B: Z = D * U
  v8f acc2 = {};
#pragma unroll
  for (int kk = 0; kk < 4; ++kk) {
    int k0 = kk * 4 + half * 2;
    v2f a, bf;
    a[0]  = dmat[q * 16 + k0];
    a[1]  = dmat[q * 16 + k0 + 1];
    bf[0] = ub[wid][(k0)     * 16 + q];
    bf[1] = ub[wid][(k0 + 1) * 16 + q];
    acc2 = wmma_f32k4(a, bf, acc2);
  }
#pragma unroll
  for (int v = 0; v < 8; ++v) {
    int hw = (v + 8 * half) * 16 + q;
    size_t n = (size_t)b * 256 + hw;
    float z = acc2[v];
    dct[n * DSTRIDE + c]      = z;
    act0[n * (size_t)C0P + c] = (__bf16)(z * (1.f / 256.f));
  }
}

// ---------------- bf16 GEMM core: TDM double-buffered B panel in LDS --------
// block = 8 waves, each wave one 16x16 tile; block covers 16M x 128N.
template <int KP>
__device__ inline v8f gemm_core(const __bf16* __restrict__ W,
                                const __bf16* __restrict__ Act,
                                int m0, int n0b, int wid, int lane,
                                __bf16* ldsB /* [2][128*128] */) {
  constexpr int NKS = KP / 128;
  int half = lane >> 4, q = lane & 15;
  const __bf16* ar = W + (size_t)(m0 + q) * KP + 16 * half;
  v8f acc = {};
  if (wid == 0)                       // one wave drives the Tensor Data Mover
    tdm_stage(ldsB, Act + (size_t)n0b * KP, KP / 4);
  for (int ks = 0; ks < NKS; ++ks) {
    int cur = ks & 1;
    if (wid == 0) {
      if (ks + 1 < NKS) {
        tdm_stage(ldsB + (cur ^ 1) * (128 * 128),
                  Act + (size_t)n0b * KP + (ks + 1) * 128, KP / 4);
        __builtin_amdgcn_s_wait_tensorcnt(1);  // oldest (buffer 'cur') arrived
      } else {
        __builtin_amdgcn_s_wait_tensorcnt(0);
      }
    }
    __syncthreads();                  // publish TDM-filled buffer to all waves
    const __bf16* arow = ar + ks * 128;
    const __bf16* brow =
        ldsB + cur * (128 * 128) + (wid * 16 + q) * 128 + 16 * half;
#pragma unroll
    for (int kk = 0; kk < 4; ++kk) {
      __builtin_prefetch(arow + kk * 32 + 512, 0, 0);
      v8bf alo = *(const v8bf*)(arow + kk * 32);
      v8bf ahi = *(const v8bf*)(arow + kk * 32 + 8);
      v16bf a;
#pragma unroll
      for (int i = 0; i < 8; ++i) { a[i] = alo[i]; a[i + 8] = ahi[i]; }
      v16bf bb = *(const v16bf*)(brow + kk * 32);
      acc = wmma_bf16x32(a, bb, acc);
    }
    __syncthreads();                  // buffer free before TDM overwrites it
  }
  return acc;
}

template <int KP>
__global__ void k_gemm_relu(const __bf16* __restrict__ W,
                            const __bf16* __restrict__ Act,
                            const float* __restrict__ bias,
                            __bf16* __restrict__ Out, int Mp, int Oreal) {
  __shared__ __bf16 ldsB[2][128 * 128];          // 64 KB
  int mt  = blockIdx.x % (Mp >> 4);
  int nb  = blockIdx.x / (Mp >> 4);
  int wid = threadIdx.x >> 5, lane = threadIdx.x & 31;
  int half = lane >> 4, q = lane & 15;
  int m0 = mt * 16, n0b = nb * 128;
  v8f acc = gemm_core<KP>(W, Act, m0, n0b, wid, lane, &ldsB[0][0]);
  int mb = m0 + 8 * half;
  int n  = n0b + wid * 16 + q;
  v8bf o;
#pragma unroll
  for (int v = 0; v < 8; ++v) {
    int mm = mb + v;
    float bv = (mm < Oreal) ? bias[mm] : 0.f;
    o[v] = (__bf16)fmaxf(acc[v] + bv, 0.f);
  }
  *(v8bf*)(Out + (size_t)n * Mp + mb) = o;
}

template <int KP>
__global__ void k_gemm_mask(const __bf16* __restrict__ W,
                            const __bf16* __restrict__ Act,
                            const float* __restrict__ bias,
                            const float* __restrict__ dct,
                            float* __restrict__ clean, int Mp, int Oreal) {
  __shared__ __bf16 ldsB[2][128 * 128];          // 64 KB
  int mt  = blockIdx.x % (Mp >> 4);
  int nb  = blockIdx.x / (Mp >> 4);
  int wid = threadIdx.x >> 5, lane = threadIdx.x & 31;
  int half = lane >> 4, q = lane & 15;
  int m0 = mt * 16, n0b = nb * 128;
  v8f acc = gemm_core<KP>(W, Act, m0, n0b, wid, lane, &ldsB[0][0]);
  int mb = m0 + 8 * half;
  int n  = n0b + wid * 16 + q;
#pragma unroll
  for (int v = 0; v < 8; ++v) {
    int mm = mb + v;
    float bv = (mm < Oreal) ? bias[mm] : 0.f;
    float s  = 1.f / (1.f + __expf(-(acc[v] + bv)));
    size_t idx = (size_t)n * DSTRIDE + mm;
    clean[idx] = s * dct[idx];
  }
}

// ---------------- inverse DCT + overlap-add (atomics) ------------------------
__global__ void k_idct(const float* __restrict__ clean,
                       const float* __restrict__ dmat,   // base; idct at +256
                       float* __restrict__ out) {
  __shared__ float tile[8][256];
  __shared__ float ub[8][256];
  const float* di = dmat + 256;
  int wid = threadIdx.x >> 5, lane = threadIdx.x & 31;
  int t = blockIdx.x * 8 + wid;
  int b = t / C0, c = t % C0;
  int ch = c % 3, gi = c / 3, gx = gi % G, gy = gi / G;
#pragma unroll
  for (int i = 0; i < 8; ++i) {
    int hw = lane * 8 + i;
    tile[wid][hw] = clean[((size_t)b * 256 + hw) * DSTRIDE + c];
  }
  __syncthreads();
  int half = lane >> 4, q = lane & 15;
  // stage A: V = Cd * Di^T
  v8f acc = {};
#pragma unroll
  for (int kk = 0; kk < 4; ++kk) {
    int k0 = kk * 4 + half * 2;
    v2f a, bf;
    a[0]  = tile[wid][q * 16 + k0];
    a[1]  = tile[wid][q * 16 + k0 + 1];
    bf[0] = di[q * 16 + k0];
    bf[1] = di[q * 16 + k0 + 1];
    acc = wmma_f32k4(a, bf, acc);
  }
#pragma unroll
  for (int v = 0; v < 8; ++v) ub[wid][(v + 8 * half) * 16 + q] = acc[v];
  __syncthreads();
  // stage B: Z = Di * V
  v8f acc2 = {};
#pragma unroll
  for (int kk = 0; kk < 4; ++kk) {
    int k0 = kk * 4 + half * 2;
    v2f a, bf;
    a[0]  = di[q * 16 + k0];
    a[1]  = di[q * 16 + k0 + 1];
    bf[0] = ub[wid][(k0)     * 16 + q];
    bf[1] = ub[wid][(k0 + 1) * 16 + q];
    acc2 = wmma_f32k4(a, bf, acc2);
  }
  float* dst = out + (((size_t)(b * 3 + ch) * IMG + gy * STR) * IMG + gx * STR);
#pragma unroll
  for (int v = 0; v < 8; ++v) {
    int mr = v + 8 * half;
    atomicAdd(dst + (size_t)mr * IMG + q, acc2[v]);
  }
}

__global__ void k_norm(float* out, size_t total) {
  size_t i = blockIdx.x * (size_t)blockDim.x + threadIdx.x;
  if (i >= total) return;
  int px = (int)(i % IMG);
  int py = (int)((i / IMG) % IMG);
  auto cnt = [](int v) {
    int gmin = (v >= 15) ? ((v - 12) >> 2) : 0;
    int gmax = v >> 2; if (gmax > G - 1) gmax = G - 1;
    return gmax - gmin + 1;
  };
  out[i] = out[i] / (float)(cnt(px) * cnt(py));
}

// ---------------- launch ----------------
extern "C" void kernel_launch(void* const* d_in, const int* in_sizes, int n_in,
                              void* d_out, int out_size, void* d_ws, size_t ws_size,
                              hipStream_t stream) {
  const float* x  = (const float*)d_in[0];
  const float* w1 = (const float*)d_in[1]; const float* b1 = (const float*)d_in[2];
  const float* w2 = (const float*)d_in[3]; const float* b2 = (const float*)d_in[4];
  const float* w3 = (const float*)d_in[5]; const float* b3 = (const float*)d_in[6];
  const float* w4 = (const float*)d_in[7]; const float* b4 = (const float*)d_in[8];
  float* out = (float*)d_out;

  // workspace carve-up (~275 MB total)
  char* ws = (char*)d_ws;
  float*  dmat = (float*)ws;                    size_t off = 4096;
  __bf16* wb1  = (__bf16*)(ws + off);           off += (size_t)C1P * C0P * 2;
  __bf16* wb2  = (__bf16*)(ws + off);           off += (size_t)C2P * C1P * 2;
  __bf16* wb3  = (__bf16*)(ws + off);           off += (size_t)C3P * C2P * 2;
  __bf16* wb4  = (__bf16*)(ws + off);           off += (size_t)C4P * C3P * 2;
  __bf16* act0 = (__bf16*)(ws + off);           off += (size_t)NCOL * C0P * 2;
  __bf16* act1 = (__bf16*)(ws + off);           off += (size_t)NCOL * C1P * 2;
  __bf16* act2 = (__bf16*)(ws + off);           off += (size_t)NCOL * C2P * 2;
  __bf16* act3 = (__bf16*)(ws + off);           off += (size_t)NCOL * C3P * 2;
  float*  dct  = (float*)(ws + off);            off += (size_t)NCOL * DSTRIDE * 4;
  float*  clean= (float*)(ws + off);            off += (size_t)NCOL * DSTRIDE * 4;

  k_init_dmat<<<1, 256, 0, stream>>>(dmat);
  k_zero_u32<<<4096, 256, 0, stream>>>((uint32_t*)act0, (size_t)NCOL * C0P / 2);
  k_zero_u32<<<4096, 256, 0, stream>>>((uint32_t*)out, (size_t)out_size);

  k_wcast<<<4096, 256, 0, stream>>>(w1, wb1, C1, C0, C0P, (size_t)C1P * C0P);
  k_wcast<<<4096, 256, 0, stream>>>(w2, wb2, C2, C1, C1P, (size_t)C2P * C1P);
  k_wcast<<<4096, 256, 0, stream>>>(w3, wb3, C3, C2, C2P, (size_t)C3P * C2P);
  k_wcast<<<4096, 256, 0, stream>>>(w4, wb4, C4, C3, C3P, (size_t)C4P * C3P);

  // forward DCT: 8*8427 tiles, 8 tiles/block -> exactly 8427 blocks
  k_dct<<<C0, 256, 0, stream>>>(x, dmat, dct, act0);

  // MLP layers (grid = (Mp/16) M-tiles x 16 N-blocks of 128 cols)
  k_gemm_relu<C0P><<<(C1P / 16) * (NCOL / 128), 256, 0, stream>>>(
      wb1, act0, b1, act1, C1P, C1);
  k_gemm_relu<C1P><<<(C2P / 16) * (NCOL / 128), 256, 0, stream>>>(
      wb2, act1, b2, act2, C2P, C2);
  k_gemm_relu<C2P><<<(C3P / 16) * (NCOL / 128), 256, 0, stream>>>(
      wb3, act2, b3, act3, C3P, C3);
  k_gemm_mask<C3P><<<(C4P / 16) * (NCOL / 128), 256, 0, stream>>>(
      wb4, act3, b4, dct, clean, C4P, C4);

  // inverse DCT + overlap-add, then normalize by coverage count
  k_idct<<<C0, 256, 0, stream>>>(clean, dmat, out);
  size_t total = (size_t)out_size;
  k_norm<<<(unsigned)((total + 255) / 256), 256, 0, stream>>>(out, total);
}
